// DARNN_22247930593894
// MI455X (gfx1250) — matrix operand order repaired and checked by reading, compiled, not run on hardware
//
#include <hip/hip_runtime.h>
#include <math.h>

// Problem dims
#define BB 256
#define TT 64
#define NN 128
#define MM 256
#define PP 256
#define IN_LD 129   // N + YD

typedef __attribute__((ext_vector_type(16))) __bf16 v16bf;
typedef __attribute__((ext_vector_type(8)))  float  v8f;
typedef unsigned short u16;

// ---------------- f32 -> bf16 (RNE) ----------------
__device__ __forceinline__ unsigned int f2bf(float f) {
  union { float f; unsigned int u; } c; c.f = f;
  return ((c.u + 0x7FFFu + ((c.u >> 16) & 1u)) >> 16) & 0xFFFFu;
}

// ---------------- bf16 WMMA fragment loaders (pure b128 loads) ----------------
// Requires: row base 16B-aligned (ld % 8 == 0, base 16B-aligned), k offsets % 32 == 0.

// A fragment 16x32 (MxK): lane L<16 -> row L, K {0..7,16..23}; lanes 16..31 -> K {8..15,24..31}
__device__ __forceinline__ v16bf load_a_frag(const u16* __restrict__ p0, int ld, int lane) {
  const int r  = lane & 15;
  const int kb = (lane >> 4) << 3;                    // 0 or 8
  const u16* p = p0 + (size_t)r * ld + kb;
  union { uint4 q[2]; v16bf v; } f;
  f.q[0] = *(const uint4*)(p);                        // K kb .. kb+7
  f.q[1] = *(const uint4*)(p + 16);                   // K kb+16 .. kb+23
  return f.v;
}

// B fragment 32x16 (KxN) = W^T tile from row-major W[n][k]:
// lane L<16 -> col L, K {0..7,16..23}; lanes 16..31 -> col L-16, K {8..15,24..31}
__device__ __forceinline__ v16bf load_b_frag(const u16* __restrict__ p0, int ld, int lane) {
  const int n  = lane & 15;
  const int kb = (lane >> 4) << 3;                    // 0 or 8
  const u16* p = p0 + (size_t)n * ld + kb;
  union { uint4 q[2]; v16bf v; } f;
  f.q[0] = *(const uint4*)(p);
  f.q[1] = *(const uint4*)(p + 16);
  return f.v;
}

// ---------------- WMMA GEMM: C = A @ W^T  (bf16 in, f32 out) ----------------
// One wave per 16x32 output strip (two 16x16 WMMA tiles, shared A fragment).
// 4 waves / 128 threads per block. K % 32 == 0, Ncols % 32 == 0.
__global__ void k_gemm(const u16* __restrict__ A, int lda,
                       const u16* __restrict__ W, int ldw,
                       float* __restrict__ C, int ldc,
                       int Mrows, int Ncols, int K) {
  const int lane   = threadIdx.x & 31;
  const int strip  = blockIdx.x * (blockDim.x >> 5) + (threadIdx.x >> 5);
  const int scols  = Ncols >> 5;
  const int nstrip = (Mrows >> 4) * scols;
  if (strip >= nstrip) return;                        // wave-uniform
  const int tm = strip / scols, tp = strip % scols;
  const u16* arow  = A + (size_t)(tm << 4) * lda;
  const u16* wrow0 = W + (size_t)(tp << 5) * ldw;
  const u16* wrow1 = wrow0 + (size_t)16 * ldw;
  v8f acc0, acc1;
#pragma unroll
  for (int i = 0; i < 8; ++i) { acc0[i] = 0.0f; acc1[i] = 0.0f; }
  for (int k = 0; k < K; k += 32) {
    v16bf af  = load_a_frag(arow + k, lda, lane);
    v16bf bf0 = load_b_frag(wrow0 + k, ldw, lane);
    v16bf bf1 = load_b_frag(wrow1 + k, ldw, lane);
    acc0 = __builtin_amdgcn_wmma_f32_16x16x32_bf16(false, af, false, bf0,
                                                   (short)0, acc0, false, false);
    acc1 = __builtin_amdgcn_wmma_f32_16x16x32_bf16(false, af, false, bf1,
                                                   (short)0, acc1, false, false);
  }
  const int row0 = (tm << 4) + ((lane >> 4) << 3);
  const int col0 = (tp << 5) + (lane & 15);
#pragma unroll
  for (int r = 0; r < 8; ++r) {
    C[(size_t)(row0 + r) * ldc + col0]      = acc0[r];
    C[(size_t)(row0 + r) * ldc + col0 + 16] = acc1[r];
  }
}

// ---------------- fused dual-GEMM + bias + rank-1 ----------------
// C = [A1@W1^T if K1>0] + [A2@W2^T if K2>0] + b1[n] + b2[n] + r1x[m]*r1w[n]
__global__ void k_gemm2(const u16* __restrict__ A1, int lda1,
                        const u16* __restrict__ W1, int ldw1, int K1,
                        const u16* __restrict__ A2, int lda2,
                        const u16* __restrict__ W2, int ldw2, int K2,
                        const float* __restrict__ b1, const float* __restrict__ b2,
                        const float* __restrict__ r1x, const float* __restrict__ r1w,
                        float* __restrict__ C, int ldc, int Mrows, int Ncols) {
  const int lane   = threadIdx.x & 31;
  const int strip  = blockIdx.x * (blockDim.x >> 5) + (threadIdx.x >> 5);
  const int scols  = Ncols >> 5;
  const int nstrip = (Mrows >> 4) * scols;
  if (strip >= nstrip) return;
  const int tm = strip / scols, tp = strip % scols;
  v8f acc0, acc1;
#pragma unroll
  for (int i = 0; i < 8; ++i) { acc0[i] = 0.0f; acc1[i] = 0.0f; }
  if (K1 > 0) {
    const u16* arow  = A1 + (size_t)(tm << 4) * lda1;
    const u16* wrow0 = W1 + (size_t)(tp << 5) * ldw1;
    const u16* wrow1 = wrow0 + (size_t)16 * ldw1;
    for (int k = 0; k < K1; k += 32) {
      v16bf af  = load_a_frag(arow + k, lda1, lane);
      v16bf bf0 = load_b_frag(wrow0 + k, ldw1, lane);
      v16bf bf1 = load_b_frag(wrow1 + k, ldw1, lane);
      acc0 = __builtin_amdgcn_wmma_f32_16x16x32_bf16(false, af, false, bf0,
                                                     (short)0, acc0, false, false);
      acc1 = __builtin_amdgcn_wmma_f32_16x16x32_bf16(false, af, false, bf1,
                                                     (short)0, acc1, false, false);
    }
  }
  if (K2 > 0) {
    const u16* arow  = A2 + (size_t)(tm << 4) * lda2;
    const u16* wrow0 = W2 + (size_t)(tp << 5) * ldw2;
    const u16* wrow1 = wrow0 + (size_t)16 * ldw2;
    for (int k = 0; k < K2; k += 32) {
      v16bf af  = load_a_frag(arow + k, lda2, lane);
      v16bf bf0 = load_b_frag(wrow0 + k, ldw2, lane);
      v16bf bf1 = load_b_frag(wrow1 + k, ldw2, lane);
      acc0 = __builtin_amdgcn_wmma_f32_16x16x32_bf16(false, af, false, bf0,
                                                     (short)0, acc0, false, false);
      acc1 = __builtin_amdgcn_wmma_f32_16x16x32_bf16(false, af, false, bf1,
                                                     (short)0, acc1, false, false);
    }
  }
  const int row0 = (tm << 4) + ((lane >> 4) << 3);
  const int col0 = (tp << 5) + (lane & 15);
  float ba0 = 0.0f, ba1 = 0.0f;
  if (b1) { ba0 += b1[col0]; ba1 += b1[col0 + 16]; }
  if (b2) { ba0 += b2[col0]; ba1 += b2[col0 + 16]; }
  const float rw0 = r1w ? r1w[col0] : 0.0f;
  const float rw1 = r1w ? r1w[col0 + 16] : 0.0f;
#pragma unroll
  for (int r = 0; r < 8; ++r) {
    float v0 = acc0[r] + ba0;
    float v1 = acc1[r] + ba1;
    if (r1x) { float x = r1x[row0 + r]; v0 += x * rw0; v1 += x * rw1; }
    C[(size_t)(row0 + r) * ldc + col0]      = v0;
    C[(size_t)(row0 + r) * ldc + col0 + 16] = v1;
  }
}

// ---------------- small elementwise / fused kernels ----------------

__global__ void k_zero(float* p, int n) {
  int i = blockIdx.x * blockDim.x + threadIdx.x;
  if (i < n) p[i] = 0.0f;
}
__global__ void k_zero16(u16* p, int n) {
  int i = blockIdx.x * blockDim.x + threadIdx.x;
  if (i < n) p[i] = 0;                                // +0.0 bf16
}
__global__ void k_f2bf(const float* __restrict__ src, u16* __restrict__ dst, int n) {
  int i = blockIdx.x * blockDim.x + threadIdx.x;
  if (i < n) dst[i] = (u16)f2bf(src[i]);
}

// X_nt[b][n][t] = inputs[b][t][n], emitted directly as bf16 (GEMM A operand only)
__global__ void k_xnt(const float* __restrict__ in, u16* __restrict__ xnt) {
  int i = blockIdx.x * blockDim.x + threadIdx.x;      // B*N*T
  int t = i & 63, n = (i >> 6) & 127, b = i >> 13;
  xnt[i] = (u16)f2bf(in[((size_t)b * TT + t) * IN_LD + n]);
}

__device__ __forceinline__ float sigf(float x) { return 1.0f / (1.0f + expf(-x)); }

// Encoder input attention; writes xa (bf16, GEMM operand). Block = b, 128 threads = n.
__global__ void k_enc_attn(const float* __restrict__ Xatt, const float* __restrict__ S,
                           const float* __restrict__ v_e, const float* __restrict__ in,
                           int tstep, u16* __restrict__ xa) {
  int b = blockIdx.x, n = threadIdx.x;
  const float* xr = Xatt + ((size_t)b * NN + n) * TT;
  const float* sr = S + (size_t)b * TT;
  float e = 0.0f;
  for (int t = 0; t < TT; ++t) e += v_e[t] * tanhf(xr[t] + sr[t]);
  __shared__ float red[NN];
  red[n] = e; __syncthreads();
  for (int s = NN >> 1; s > 0; s >>= 1) { if (n < s) red[n] = fmaxf(red[n], red[n + s]); __syncthreads(); }
  float mx = red[0]; __syncthreads();
  float ex = expf(e - mx);
  red[n] = ex; __syncthreads();
  for (int s = NN >> 1; s > 0; s >>= 1) { if (n < s) red[n] += red[n + s]; __syncthreads(); }
  float alpha = ex / red[0];
  xa[(size_t)b * NN + n] = (u16)f2bf(in[((size_t)b * TT + tstep) * IN_LD + n] * alpha);
}

// Decoder temporal attention + context + y_tilde. Block = b, 64 threads = t.
__global__ void k_dec_attn(const float* __restrict__ XattD, const float* __restrict__ Sd,
                           const float* __restrict__ v_d, const float* __restrict__ Xe,
                           const float* __restrict__ in, int tstep,
                           const float* __restrict__ wbt,
                           float* __restrict__ ctx, float* __restrict__ ytil) {
  int b = blockIdx.x, t = threadIdx.x;
  const float* xr = XattD + ((size_t)b * TT + t) * MM;
  const float* sr = Sd + (size_t)b * MM;
  float l = 0.0f;
  for (int m = 0; m < MM; ++m) l += v_d[m] * tanhf(xr[m] + sr[m]);
  __shared__ float red[TT];
  __shared__ float sbeta[TT];
  red[t] = l; __syncthreads();
  for (int s = TT >> 1; s > 0; s >>= 1) { if (t < s) red[t] = fmaxf(red[t], red[t + s]); __syncthreads(); }
  float mx = red[0]; __syncthreads();
  float ex = expf(l - mx);
  red[t] = ex; __syncthreads();
  for (int s = TT >> 1; s > 0; s >>= 1) { if (t < s) red[t] += red[t + s]; __syncthreads(); }
  float beta = ex / red[0]; __syncthreads();
  sbeta[t] = beta; __syncthreads();
  float part = 0.0f;
#pragma unroll
  for (int j = 0; j < 4; ++j) {
    int m = t * 4 + j;
    float acc = 0.0f;
    for (int tt = 0; tt < TT; ++tt) acc += sbeta[tt] * Xe[((size_t)b * TT + tt) * MM + m];
    ctx[(size_t)b * MM + m] = acc;
    part += acc * wbt[1 + m];
  }
  red[t] = part; __syncthreads();
  for (int s = TT >> 1; s > 0; s >>= 1) { if (t < s) red[t] += red[t + s]; __syncthreads(); }
  if (t == 0)
    ytil[b] = in[((size_t)b * TT + tstep) * IN_LD + NN] * wbt[0] + red[0];
}

// LSTM cell. gates [B,1024] = i|f|g|o. hc = [h(256)|c(256)] f32; hc_bf mirrored bf16.
// Optionally mirrors h into Xe (f32) and Xe_bf (bf16) at time tstep.
__global__ void k_lstm_cell(const float* __restrict__ gates, float* __restrict__ hc,
                            u16* __restrict__ hc_bf,
                            float* __restrict__ xe, u16* __restrict__ xe_bf, int tstep) {
  int i = blockIdx.x * blockDim.x + threadIdx.x;      // B*256
  int b = i >> 8, j = i & 255;
  const float* g = gates + (size_t)b * 1024;
  float gi = sigf(g[j]);
  float gf = sigf(g[256 + j]);
  float gg = tanhf(g[512 + j]);
  float go = sigf(g[768 + j]);
  float c  = hc[(size_t)b * 512 + 256 + j];
  float cn = gf * c + gi * gg;
  float hn = go * tanhf(cn);
  hc[(size_t)b * 512 + j]          = hn;
  hc[(size_t)b * 512 + 256 + j]    = cn;
  hc_bf[(size_t)b * 512 + j]       = (u16)f2bf(hn);
  hc_bf[(size_t)b * 512 + 256 + j] = (u16)f2bf(cn);
  if (xe) {
    xe[((size_t)b * TT + tstep) * MM + j]    = hn;
    xe_bf[((size_t)b * TT + tstep) * MM + j] = (u16)f2bf(hn);
  }
}

// hctx_bf = bf16(concat(h_d, ctx))  [B,512]
__global__ void k_build_hctx(const float* __restrict__ hcd, const float* __restrict__ ctx,
                             u16* __restrict__ hctx) {
  int i = blockIdx.x * blockDim.x + threadIdx.x;      // B*512
  int b = i >> 9, j = i & 511;
  float v = (j < 256) ? hcd[(size_t)b * 512 + j] : ctx[(size_t)b * 256 + (j - 256)];
  hctx[i] = (u16)f2bf(v);
}

__global__ void k_final(const float* __restrict__ hidden, const float* __restrict__ vbW,
                        const float* __restrict__ vbb, float* __restrict__ out) {
  int b = threadIdx.x;                                // one block of 256
  float acc = vbb[0];
  for (int m = 0; m < PP; ++m) acc += hidden[(size_t)b * PP + m] * vbW[m];
  out[b] = acc;
}

// ---------------- host side ----------------

extern "C" void kernel_launch(void* const* d_in, const int* in_sizes, int n_in,
                              void* d_out, int out_size, void* d_ws, size_t ws_size,
                              hipStream_t stream) {
  const float* inputs   = (const float*)d_in[0];
  const float* WU_e     = (const float*)d_in[1];
  const float* v_e      = (const float*)d_in[2];
  const float* Wih_e    = (const float*)d_in[3];
  const float* Whh_e    = (const float*)d_in[4];
  const float* bih_e    = (const float*)d_in[5];
  const float* bhh_e    = (const float*)d_in[6];
  const float* WU_d     = (const float*)d_in[7];
  const float* v_d      = (const float*)d_in[8];
  const float* wb_tilde = (const float*)d_in[9];
  const float* Wih_d    = (const float*)d_in[10];
  const float* Whh_d    = (const float*)d_in[11];
  const float* bih_d    = (const float*)d_in[12];
  const float* bhh_d    = (const float*)d_in[13];
  const float* Wb_W     = (const float*)d_in[14];
  const float* Wb_b     = (const float*)d_in[15];
  const float* vb_W     = (const float*)d_in[16];
  const float* vb_b     = (const float*)d_in[17];
  float* out = (float*)d_out;

  // f32 scratch (front), bf16 scratch (back), all 256B-aligned
  char* base = (char*)d_ws;
  size_t off = 0;
  auto allocf = [&](size_t n) { float* p = (float*)(base + off); off += ((n * 4 + 255) & ~(size_t)255); return p; };
  auto allocb = [&](size_t n) { u16* p = (u16*)(base + off); off += ((n * 2 + 255) & ~(size_t)255); return p; };

  float* xatt   = allocf((size_t)BB * NN * TT);   // 8 MB
  float* xe     = allocf((size_t)BB * TT * MM);   // 16 MB
  float* xattd  = allocf((size_t)BB * TT * MM);   // 16 MB
  float* S      = allocf((size_t)BB * TT);
  float* Sd     = allocf((size_t)BB * MM);
  float* hc     = allocf((size_t)BB * 512);
  float* hcd    = allocf((size_t)BB * 512);
  float* gates  = allocf((size_t)BB * 1024);
  float* ctx    = allocf((size_t)BB * MM);
  float* ytil   = allocf((size_t)BB);
  float* hidden = allocf((size_t)BB * PP);

  u16* xnt_b   = allocb((size_t)BB * NN * TT);    // 4 MB
  u16* xe_b    = allocb((size_t)BB * TT * MM);    // 8 MB
  u16* xa_b    = allocb((size_t)BB * NN);
  u16* hc_b    = allocb((size_t)BB * 512);
  u16* hcd_b   = allocb((size_t)BB * 512);
  u16* hctx_b  = allocb((size_t)BB * 512);
  u16* wWUe    = allocb((size_t)TT * (2 * MM + TT));     // [64,576]
  u16* wWihe   = allocb((size_t)4 * MM * NN);            // [1024,128]
  u16* wWhhe   = allocb((size_t)4 * MM * MM);            // [1024,256]
  u16* wWUd    = allocb((size_t)MM * (2 * PP + MM));     // [256,768]
  u16* wWhhd   = allocb((size_t)4 * PP * PP);            // [1024,256]
  u16* wWbW    = allocb((size_t)PP * (PP + MM));         // [256,512]
  if (off > ws_size) return;                              // workspace too small

  auto cvt = [&](const float* s, u16* d, int n) {
    k_f2bf<<<dim3((n + 255) / 256), dim3(256), 0, stream>>>(s, d, n);
  };
  auto gemm = [&](const u16* A, int lda, const u16* W, int ldw,
                  float* C, int ldc, int M_, int N_, int K_) {
    int strips = (M_ >> 4) * (N_ >> 5);
    k_gemm<<<dim3((strips + 3) / 4), dim3(128), 0, stream>>>(A, lda, W, ldw, C, ldc, M_, N_, K_);
  };
  auto gemm2 = [&](const u16* A1, int lda1, const u16* W1, int ldw1, int K1,
                   const u16* A2, int lda2, const u16* W2, int ldw2, int K2,
                   const float* b1, const float* b2,
                   const float* r1x, const float* r1w,
                   float* C, int ldc, int M_, int N_) {
    int strips = (M_ >> 4) * (N_ >> 5);
    k_gemm2<<<dim3((strips + 3) / 4), dim3(128), 0, stream>>>(
        A1, lda1, W1, ldw1, K1, A2, lda2, W2, ldw2, K2, b1, b2, r1x, r1w, C, ldc, M_, N_);
  };

  // one-time weight conversions (L2-resident afterwards)
  cvt(WU_e,  wWUe,  TT * (2 * MM + TT));
  cvt(Wih_e, wWihe, 4 * MM * NN);
  cvt(Whh_e, wWhhe, 4 * MM * MM);
  cvt(WU_d,  wWUd,  MM * (2 * PP + MM));
  cvt(Whh_d, wWhhd, 4 * PP * PP);
  cvt(Wb_W,  wWbW,  PP * (PP + MM));

  // init states (f32 + bf16 mirrors)
  k_zero<<<dim3(BB * 512 / 256), dim3(256), 0, stream>>>(hc,  BB * 512);
  k_zero<<<dim3(BB * 512 / 256), dim3(256), 0, stream>>>(hcd, BB * 512);
  k_zero16<<<dim3(BB * 512 / 256), dim3(256), 0, stream>>>(hc_b,  BB * 512);
  k_zero16<<<dim3(BB * 512 / 256), dim3(256), 0, stream>>>(hcd_b, BB * 512);

  // X_nt (bf16) + hoisted time-invariant encoder attention term:
  // xatt = X_nt @ WU_e[:,2M:].T   (bf16 WU_e subview: ld 576, col offset 512)
  k_xnt<<<dim3(BB * NN * TT / 256), dim3(256), 0, stream>>>(inputs, xnt_b);
  gemm(xnt_b, TT, wWUe + 2 * MM, 2 * MM + TT, xatt, TT, BB * NN, TT, TT);

  // -------- encoder scan (T steps) --------
  for (int t = 0; t < TT; ++t) {
    gemm(hc_b, 512, wWUe, 2 * MM + TT, S, TT, BB, TT, 2 * MM);          // S = [h|c]@WU_e_h^T
    k_enc_attn<<<dim3(BB), dim3(NN), 0, stream>>>(xatt, S, v_e, inputs, t, xa_b);
    gemm2(xa_b, NN, wWihe, NN, NN, hc_b, 512, wWhhe, MM, MM,
          bih_e, bhh_e, nullptr, nullptr, gates, 1024, BB, 1024);
    k_lstm_cell<<<dim3(BB * MM / 256), dim3(256), 0, stream>>>(gates, hc, hc_b, xe, xe_b, t);
  }

  // hoisted decoder attention term: xattd = Xe @ WU_d[:,2P:].T
  gemm(xe_b, MM, wWUd + 2 * PP, 2 * PP + MM, xattd, MM, BB * TT, MM, MM);

  // -------- decoder scan (T-1 steps) --------
  for (int s = 0; s < TT - 1; ++s) {
    gemm(hcd_b, 512, wWUd, 2 * PP + MM, Sd, MM, BB, MM, 2 * PP);
    k_dec_attn<<<dim3(BB), dim3(TT), 0, stream>>>(xattd, Sd, v_d, xe, inputs, s,
                                                  wb_tilde, ctx, ytil);
    gemm2(nullptr, 0, nullptr, 0, 0, hcd_b, 512, wWhhd, PP, PP,
          bih_d, bhh_d, ytil, Wih_d, gates, 1024, BB, 1024);   // rank-1 y_tilde*Wih_d
    k_lstm_cell<<<dim3(BB * PP / 256), dim3(256), 0, stream>>>(gates, hcd, hcd_b,
                                                               nullptr, nullptr, 0);
  }

  // -------- output head --------
  k_build_hctx<<<dim3(BB * 512 / 256), dim3(256), 0, stream>>>(hcd, ctx, hctx_b);
  gemm2(hctx_b, 512, wWbW, PP + MM, PP + MM, nullptr, 0, nullptr, 0, 0,
        Wb_b, nullptr, nullptr, nullptr, hidden, PP, BB, PP);
  k_final<<<dim3(1), dim3(BB), 0, stream>>>(hidden, vb_W, vb_b, out);
}
